// GCN_3_52561809768789
// MI455X (gfx1250) — matrix operand order repaired and checked by reading, not compiled, not run on hardware
//
#include <hip/hip_runtime.h>
#include <hip/hip_bf16.h>

typedef __bf16 bf16;
typedef __attribute__((ext_vector_type(16))) __bf16 v16bf;
typedef __attribute__((ext_vector_type(8)))  __bf16 v8bf;
typedef __attribute__((ext_vector_type(8)))  float  v8f;

#define BATCH 4
#define NODES 1568
#define CH    2048
#define LSTR  40   // LDS row stride in bf16 (32 data + 8 pad = 80B; keeps 16B chunk alignment)
#define TM    128
#define TN    128

// ---------------------------------------------------------------- conversions

__global__ __launch_bounds__(256) void k_cvt(const float* __restrict__ s, bf16* __restrict__ d, long n) {
  long stride = (long)gridDim.x * blockDim.x;
  for (long i = (long)blockIdx.x * blockDim.x + threadIdx.x; i < n; i += stride)
    d[i] = (bf16)s[i];
}

// x [b][n][c] -> xT [b][c][n]  (bf16)
__global__ __launch_bounds__(256) void k_cvtXT(const float* __restrict__ x, bf16* __restrict__ xT) {
  long n = (long)BATCH * NODES * CH;
  long stride = (long)gridDim.x * blockDim.x;
  for (long i = (long)blockIdx.x * blockDim.x + threadIdx.x; i < n; i += stride) {
    int c = (int)(i % CH); long r = i / CH;
    int nn = (int)(r % NODES); int b = (int)(r / NODES);
    xT[((long)b * CH + c) * NODES + nn] = (bf16)x[i];
  }
}

// deconv_w [i][o][kh][kw] -> WhrT[cls][o][i], cls = kh*3+kw  (trans layout: K=i contiguous)
__global__ __launch_bounds__(256) void k_cvt_deconv(const float* __restrict__ w, bf16* __restrict__ o) {
  long n = 9L * CH * 1024;
  long stride = (long)gridDim.x * blockDim.x;
  for (long i = (long)blockIdx.x * blockDim.x + threadIdx.x; i < n; i += stride) {
    long cls = i % 9, io = i / 9;
    long ii = io / 1024, oo = io % 1024;
    o[(cls * 1024 + oo) * CH + ii] = (bf16)w[i];
  }
}

// ---------------------------------------------------------------- WMMA fragments
// A (16x32 bf16): lane l -> M = l&15; contiguous K run [hi*16, +16)
// B (32x16 bf16): lane l -> N = l&15; contiguous K run [hi*16, +16)
// C/D (16x16 f32): VGPR r -> M = r + 8*hi, N = l&15

__device__ inline v16bf ld_frag_a(const bf16* As, int m, int hi) {
  const bf16* p = As + m * LSTR + hi * 16;
  v8bf lo  = *reinterpret_cast<const v8bf*>(p);
  v8bf hi8 = *reinterpret_cast<const v8bf*>(p + 8);
  return __builtin_shufflevector(lo, hi8, 0, 1, 2, 3, 4, 5, 6, 7, 8, 9, 10, 11, 12, 13, 14, 15);
}

__device__ inline v16bf ld_frag_b(const bf16* Bs, int n, int hi) {
  const bf16* p = Bs + n * LSTR + hi * 16;
  v8bf lo  = *reinterpret_cast<const v8bf*>(p);
  v8bf hi8 = *reinterpret_cast<const v8bf*>(p + 8);
  return __builtin_shufflevector(lo, hi8, 0, 1, 2, 3, 4, 5, 6, 7, 8, 9, 10, 11, 12, 13, 14, 15);
}

#define WMMA_BF16(a, b, c) \
  __builtin_amdgcn_wmma_f32_16x16x32_bf16(false, (a), false, (b), (short)0, (c), false, false)

// Async copy of one 16B chunk, global -> LDS (ASYNCcnt-tracked; ISA 07/08).
__device__ inline void async_cp16(const bf16* gp, const bf16* lp) {
  unsigned lds = (unsigned)(size_t)lp;  // flat addr[31:0] == LDS byte address
  asm volatile("global_load_async_to_lds_b128 %0, %1, off"
               :: "v"(lds), "v"((unsigned long long)(size_t)gp) : "memory");
}

// ---------------------------------------------------------------- generic GEMM
// C[M,N] = A[M,K]*B[K,N]; A row-major bf16; B stored K-contiguous per column:
// B[k][n] = Bm[n*ldb + k]. 128x128 tile, 256 threads, wave = 32x64 (2x4 WMMA).
// Full tiles: double-buffered async global->LDS. Edge tiles: register staging.
template <bool OUT_BF16>
__global__ __launch_bounds__(256) void k_gemm(const bf16* __restrict__ A, long aBS, int lda,
                                              const bf16* __restrict__ Bm, long bBS, int ldb,
                                              void* __restrict__ Cout, long cBS, int ldc,
                                              int M, int N, int K) {
  __shared__ bf16 As[2][TM * LSTR];
  __shared__ bf16 Bs[2][TN * LSTR];
  const int tid = threadIdx.x;
  const int mbase = blockIdx.y * TM, nbase = blockIdx.x * TN;
  const bf16* Ab = A + (long)blockIdx.z * aBS;
  const bf16* Bb = Bm + (long)blockIdx.z * bBS;
  const int lane = tid & 31, wv = tid >> 5;
  const int wm = (wv >> 1) * 32, wn = (wv & 1) * 64;
  const int l15 = lane & 15, hi = lane >> 4;
  const bool full = (mbase + TM <= M) && (nbase + TN <= N);
  v8f acc[2][4] = {};

  auto compute = [&](int buf) {
    v16bf a0 = ld_frag_a(As[buf], wm + l15, hi);
    v16bf a1 = ld_frag_a(As[buf], wm + 16 + l15, hi);
#pragma unroll
    for (int ni = 0; ni < 4; ni++) {
      v16bf bfr = ld_frag_b(Bs[buf], wn + ni * 16 + l15, hi);
      acc[0][ni] = WMMA_BF16(a0, bfr, acc[0][ni]);
      acc[1][ni] = WMMA_BF16(a1, bfr, acc[1][ni]);
    }
  };

  if (full) {
    // 512 16B chunks per tile per operand; 2 A + 2 B async ops per thread (4/wave).
    auto issue = [&](int buf, int kt) {
#pragma unroll
      for (int t = 0; t < 2; t++) {
        int c = tid + t * 256, row = c >> 2, q = c & 3;
        async_cp16(Ab + (long)(mbase + row) * lda + kt + q * 8, &As[buf][row * LSTR + q * 8]);
        async_cp16(Bb + (long)(nbase + row) * ldb + kt + q * 8, &Bs[buf][row * LSTR + q * 8]);
      }
    };
    issue(0, 0);
    int buf = 0;
    for (int kt = 0; kt < K; kt += 32) {
      if (kt + 32 < K) {
        issue(buf ^ 1, kt + 32);
        asm volatile("s_wait_asynccnt 0x4" ::: "memory");  // current slab done, next in flight
      } else {
        asm volatile("s_wait_asynccnt 0x0" ::: "memory");
      }
      __syncthreads();
      compute(buf);
      __syncthreads();
      buf ^= 1;
    }
  } else {
    unsigned va[8], vb[8];
    auto loadA = [&](int kt) {
#pragma unroll
      for (int j = 0; j < 8; j++) {  // clamp + select: no exec-mask branches
        int id = tid + j * 256, row = id >> 4, kp = id & 15;
        int gr = mbase + row;
        unsigned v = *reinterpret_cast<const unsigned*>(Ab + (long)(gr < M ? gr : 0) * lda + kt + 2 * kp);
        va[j] = gr < M ? v : 0u;
      }
    };
    auto loadB = [&](int kt) {
#pragma unroll
      for (int j = 0; j < 8; j++) {
        int id = tid + j * 256, n = id >> 4, kp = id & 15;
        int gn = nbase + n;
        unsigned v = *reinterpret_cast<const unsigned*>(Bb + (long)(gn < N ? gn : 0) * ldb + kt + 2 * kp);
        vb[j] = gn < N ? v : 0u;
      }
    };
    auto stage = [&]() {
#pragma unroll
      for (int j = 0; j < 8; j++) {
        int id = tid + j * 256, row = id >> 4, kp = id & 15;
        *reinterpret_cast<unsigned*>(&As[0][row * LSTR + 2 * kp]) = va[j];
      }
#pragma unroll
      for (int j = 0; j < 8; j++) {
        int id = tid + j * 256, n = id >> 4, kp = id & 15;
        *reinterpret_cast<unsigned*>(&Bs[0][n * LSTR + 2 * kp]) = vb[j];
      }
    };
    loadA(0); loadB(0);
    for (int kt = 0; kt < K; kt += 32) {
      stage();
      __syncthreads();
      if (kt + 32 < K) { loadA(kt + 32); loadB(kt + 32); }
      compute(0);
      __syncthreads();
    }
  }

  long cOff = (long)blockIdx.z * cBS;
  if (full) {
#pragma unroll
    for (int mi = 0; mi < 2; mi++)
#pragma unroll
      for (int ni = 0; ni < 4; ni++) {
        int gn = nbase + wn + ni * 16 + l15;
#pragma unroll
        for (int r = 0; r < 8; r++) {
          int gm = mbase + wm + mi * 16 + hi * 8 + r;
          if (OUT_BF16) ((bf16*)Cout)[cOff + (long)gm * ldc + gn] = (bf16)acc[mi][ni][r];
          else          ((float*)Cout)[cOff + (long)gm * ldc + gn] = acc[mi][ni][r];
        }
      }
  } else {
#pragma unroll
    for (int mi = 0; mi < 2; mi++)
#pragma unroll
      for (int ni = 0; ni < 4; ni++) {
        int gn = nbase + wn + ni * 16 + l15;
#pragma unroll
        for (int r = 0; r < 8; r++) {
          int gm = mbase + wm + mi * 16 + hi * 8 + r;
          if (gm < M && gn < N) {
            if (OUT_BF16) ((bf16*)Cout)[cOff + (long)gm * ldc + gn] = (bf16)acc[mi][ni][r];
            else          ((float*)Cout)[cOff + (long)gm * ldc + gn] = acc[mi][ni][r];
          }
        }
      }
  }
}

// ---------------------------------------------------------------- adjacency / laplacian

__global__ __launch_bounds__(256) void k_colsum(const float* __restrict__ s, float* __restrict__ cs) {
  int b = blockIdx.y, m = blockIdx.x * 256 + threadIdx.x;
  if (m >= NODES) return;
  const float* sp = s + (long)b * NODES * NODES;
  float acc = 0.f;
  for (int n = 0; n < NODES; n++) { float t = sp[(long)n * NODES + m]; acc += t * t; }
  cs[b * NODES + m] = acc;
}

__global__ __launch_bounds__(256) void k_rowd(const float* __restrict__ s, const float* __restrict__ cs,
                                              float* __restrict__ dv) {
  __shared__ float red[256];
  int b = blockIdx.y, n = blockIdx.x;
  const float* sp = s + ((long)b * NODES + n) * NODES;
  const float* cp = cs + b * NODES;
  float acc = 0.f;
  for (int m = threadIdx.x; m < NODES; m += 256) { float t = sp[m]; acc += t * t / cp[m]; }
  red[threadIdx.x] = acc; __syncthreads();
  for (int o = 128; o > 0; o >>= 1) {
    if (threadIdx.x < o) red[threadIdx.x] += red[threadIdx.x + o];
    __syncthreads();
  }
  if (threadIdx.x == 0) dv[b * NODES + n] = rsqrtf(red[0] + 1.0f);  // +1 from identity diag
}

__global__ __launch_bounds__(256) void k_lap(const float* __restrict__ s, const float* __restrict__ cs,
                                             const float* __restrict__ dv, bf16* __restrict__ L) {
  long total = (long)BATCH * NODES * NODES;
  long stride = (long)gridDim.x * 256;
  for (long i = (long)blockIdx.x * 256 + threadIdx.x; i < total; i += stride) {
    long b = i / ((long)NODES * NODES), r = i % ((long)NODES * NODES);
    int n = (int)(r / NODES), m = (int)(r % NODES);
    float t = s[i];
    float a = t * t / cs[b * NODES + m] + (n == m ? 1.f : 0.f);
    L[i] = (bf16)(dv[b * NODES + n] * a * dv[b * NODES + m]);
  }
}

// ---------------------------------------------------------------- LayerNorm + LeakyReLU
// f32 in-place; bf16 copy written TRANSPOSED [b][c][n] for the next lap@cur GEMM.
__global__ __launch_bounds__(256) void k_ln_leaky(float* __restrict__ y, const float* __restrict__ gma,
                                                  const float* __restrict__ bta, bf16* __restrict__ ohT) {
  __shared__ float r1[256], r2[256];
  long row = blockIdx.x;
  int bb = (int)(row / NODES), nn = (int)(row % NODES);
  float* p = y + row * (long)CH;
  float v[8]; float s = 0.f, s2 = 0.f;
#pragma unroll
  for (int j = 0; j < 8; j++) { float t = p[threadIdx.x + j * 256]; v[j] = t; s += t; s2 += t * t; }
  r1[threadIdx.x] = s; r2[threadIdx.x] = s2; __syncthreads();
  for (int o = 128; o > 0; o >>= 1) {
    if (threadIdx.x < o) { r1[threadIdx.x] += r1[threadIdx.x + o]; r2[threadIdx.x] += r2[threadIdx.x + o]; }
    __syncthreads();
  }
  float mean = r1[0] * (1.f / CH);
  float var = r2[0] * (1.f / CH) - mean * mean;
  float inv = rsqrtf(var + 1e-5f);
#pragma unroll
  for (int j = 0; j < 8; j++) {
    int c = threadIdx.x + j * 256;
    float t = (v[j] - mean) * inv * gma[c] + bta[c];
    t = t >= 0.f ? t : 0.1f * t;
    p[c] = t;
    ohT[((long)bb * CH + c) * NODES + nn] = (bf16)t;
  }
}

// ---------------------------------------------------------------- head

__global__ __launch_bounds__(256) void k_conv1(const float* __restrict__ y, const float* __restrict__ cw,
                                               const float* __restrict__ cb, float* __restrict__ P) {
  long i = (long)blockIdx.x * 256 + threadIdx.x;  // 4*196*2048
  long b = i / (196L * CH), r = i % (196L * CH);
  int hw = (int)(r / CH), c = (int)(r % CH);
  float acc = cb[0];
#pragma unroll
  for (int f = 0; f < 8; f++)
    acc += y[((long)b * NODES + f * 196 + hw) * CH + c] * cw[f];
  P[((long)b * CH + c) * 196 + hw] = acc;
}

__global__ __launch_bounds__(256) void k_upsample(const float* __restrict__ P, bf16* __restrict__ U) {
  long i = (long)blockIdx.x * 256 + threadIdx.x;  // 4*28*28*2048
  int c = (int)(i % CH); long r = i / CH;
  int xx = (int)(r % 28); r /= 28;
  int yy = (int)(r % 28); int b = (int)(r / 28);
  float sy = yy * 0.5f - 0.25f, sx = xx * 0.5f - 0.25f;
  int y0 = (int)floorf(sy), x0 = (int)floorf(sx);
  float wy = sy - y0, wx = sx - x0;
  int y0c = min(13, max(0, y0)), y1c = min(13, max(0, y0 + 1));
  int x0c = min(13, max(0, x0)), x1c = min(13, max(0, x0 + 1));
  const float* p = P + ((long)b * CH + c) * 196;
  float v = (1.f - wy) * ((1.f - wx) * p[y0c * 14 + x0c] + wx * p[y0c * 14 + x1c])
          + wy        * ((1.f - wx) * p[y1c * 14 + x0c] + wx * p[y1c * 14 + x1c]);
  U[((long)(b * 28 + yy) * 28 + xx) * CH + c] = (bf16)v;
}

// ConvTranspose2d(2048->1024,k3,pad1,stride3) as 9 GEMMs (one per (kh,kw) class).
__global__ __launch_bounds__(256) void k_deconv(const bf16* __restrict__ U, const bf16* __restrict__ W,
                                                const float* __restrict__ db, float* __restrict__ out) {
  __shared__ bf16 As[2][TM * LSTR];
  __shared__ bf16 Bs[2][TN * LSTR];
  const int cls = blockIdx.z, kh = cls / 3, kw = cls % 3;
  const int nyc = (kh == 1) ? 28 : 27, nxc = (kw == 1) ? 28 : 27;
  const int h0 = (kh == 0) ? 1 : 0, w0 = (kw == 0) ? 1 : 0;
  const int R = BATCH * nyc * nxc;
  const int mbase = blockIdx.y * TM, nbase = blockIdx.x * TN;
  if (mbase >= R) return;
  const bf16* Bp = W + (long)cls * 1024 * CH;  // WhrT[cls][o][i]
  const int tid = threadIdx.x, lane = tid & 31, wv = tid >> 5;
  const int wm = (wv >> 1) * 32, wn = (wv & 1) * 64, l15 = lane & 15, hi = lane >> 4;
  const bool full = (mbase + TM <= R);  // N = 1024, always full
  v8f acc[2][4] = {};

  auto compute = [&](int buf) {
    v16bf a0 = ld_frag_a(As[buf], wm + l15, hi);
    v16bf a1 = ld_frag_a(As[buf], wm + 16 + l15, hi);
#pragma unroll
    for (int ni = 0; ni < 4; ni++) {
      v16bf bfr = ld_frag_b(Bs[buf], wn + ni * 16 + l15, hi);
      acc[0][ni] = WMMA_BF16(a0, bfr, acc[0][ni]);
      acc[1][ni] = WMMA_BF16(a1, bfr, acc[1][ni]);
    }
  };
  auto arow = [&](int r) -> const bf16* {  // gathered A row base
    int b = r / (nyc * nxc), rr = r % (nyc * nxc);
    int hh = h0 + rr / nxc, ww = w0 + rr % nxc;
    return U + (((long)b * 28 + hh) * 28 + ww) * CH;
  };

  if (full) {
    auto issue = [&](int buf, int kt) {
#pragma unroll
      for (int t = 0; t < 2; t++) {
        int c = tid + t * 256, row = c >> 2, q = c & 3;
        async_cp16(arow(mbase + row) + kt + q * 8, &As[buf][row * LSTR + q * 8]);
        async_cp16(Bp + (long)(nbase + row) * CH + kt + q * 8, &Bs[buf][row * LSTR + q * 8]);
      }
    };
    issue(0, 0);
    int buf = 0;
    for (int kt = 0; kt < CH; kt += 32) {
      if (kt + 32 < CH) {
        issue(buf ^ 1, kt + 32);
        asm volatile("s_wait_asynccnt 0x4" ::: "memory");
      } else {
        asm volatile("s_wait_asynccnt 0x0" ::: "memory");
      }
      __syncthreads();
      compute(buf);
      __syncthreads();
      buf ^= 1;
    }
  } else {
    unsigned va[8], vb[8];
    auto loadA = [&](int kt) {
#pragma unroll
      for (int j = 0; j < 8; j++) {
        int id = tid + j * 256, row = id >> 4, kp = id & 15;
        int r = mbase + row;
        unsigned v = *reinterpret_cast<const unsigned*>(arow(r < R ? r : 0) + kt + 2 * kp);
        va[j] = r < R ? v : 0u;
      }
    };
    auto loadB = [&](int kt) {
#pragma unroll
      for (int j = 0; j < 8; j++) {
        int id = tid + j * 256, n = id >> 4, kp = id & 15;
        vb[j] = *reinterpret_cast<const unsigned*>(Bp + (long)(nbase + n) * CH + kt + 2 * kp);
      }
    };
    auto stage = [&]() {
#pragma unroll
      for (int j = 0; j < 8; j++) {
        int id = tid + j * 256, row = id >> 4, kp = id & 15;
        *reinterpret_cast<unsigned*>(&As[0][row * LSTR + 2 * kp]) = va[j];
      }
#pragma unroll
      for (int j = 0; j < 8; j++) {
        int id = tid + j * 256, n = id >> 4, kp = id & 15;
        *reinterpret_cast<unsigned*>(&Bs[0][n * LSTR + 2 * kp]) = vb[j];
      }
    };
    loadA(0); loadB(0);
    for (int kt = 0; kt < CH; kt += 32) {
      stage();
      __syncthreads();
      if (kt + 32 < CH) { loadA(kt + 32); loadB(kt + 32); }
      compute(0);
      __syncthreads();
    }
  }

#pragma unroll
  for (int mi = 0; mi < 2; mi++)
#pragma unroll
    for (int ni = 0; ni < 4; ni++) {
      int o = nbase + wn + ni * 16 + l15;
      float bias = db[o];
#pragma unroll
      for (int r8 = 0; r8 < 8; r8++) {
        int gm = mbase + wm + mi * 16 + hi * 8 + r8;
        if (full || gm < R) {
          int b = gm / (nyc * nxc), rr = gm % (nyc * nxc);
          int yy = 3 * (h0 + rr / nxc) + kh - 1;
          int xx = 3 * (w0 + rr % nxc) + kw - 1;
          out[(((long)b * 1024 + o) * 82 + yy) * 82 + xx] = acc[mi][ni][r8] + bias;
        }
      }
    }
}

// ---------------------------------------------------------------- driver

extern "C" void kernel_launch(void* const* d_in, const int* in_sizes, int n_in,
                              void* d_out, int out_size, void* d_ws, size_t ws_size,
                              hipStream_t stream) {
  (void)in_sizes; (void)n_in; (void)out_size; (void)ws_size;
  const float* x     = (const float*)d_in[0];
  const float* w_fc1 = (const float*)d_in[1];
  const float* w1 = (const float*)d_in[2];  const float* g1 = (const float*)d_in[3];  const float* b1 = (const float*)d_in[4];
  const float* w2 = (const float*)d_in[5];  const float* g2 = (const float*)d_in[6];  const float* b2 = (const float*)d_in[7];
  const float* w3 = (const float*)d_in[8];  const float* g3 = (const float*)d_in[9];  const float* b3 = (const float*)d_in[10];
  const float* cw = (const float*)d_in[11]; const float* cb = (const float*)d_in[12];
  const float* dw = (const float*)d_in[13]; const float* db = (const float*)d_in[14];

  char* ws = (char*)d_ws;
  size_t off = 0;
  auto alloc = [&](size_t bytes) -> void* {
    void* p = ws + off;
    off = (off + bytes + 255) & ~(size_t)255;
    return p;
  };

  const long BNC = (long)BATCH * NODES * CH;
  const long BNN = (long)BATCH * NODES * NODES;

  bf16*  xh   = (bf16*)alloc(BNC * 2);             // x row-major bf16 (A of fc1)
  bf16*  xT   = (bf16*)alloc(BNC * 2);             // x transposed [b][c][n] (B of layer-1 lap@x)
  bf16*  wfcb = (bf16*)alloc((long)CH * CH * 2);   // weights kept [d][c]: K-contiguous per column
  bf16*  w1b  = (bf16*)alloc((long)CH * CH * 2);
  bf16*  w2b  = (bf16*)alloc((long)CH * CH * 2);
  bf16*  w3b  = (bf16*)alloc((long)CH * CH * 2);
  bf16*  WhrT = (bf16*)alloc(9L * CH * 1024 * 2);  // [cls][o][i]
  bf16*  hbuf = (bf16*)alloc(BNC * 2);
  float* sbuf = (float*)alloc(BNN * 4);
  float* cs   = (float*)alloc((long)BATCH * NODES * 4);
  float* dv   = (float*)alloc((long)BATCH * NODES * 4);
  bf16*  laph = (bf16*)alloc(BNN * 2);
  bf16*  gh   = (bf16*)alloc(BNC * 2);
  float* yf   = (float*)alloc(BNC * 4);
  bf16*  cT1  = (bf16*)alloc(BNC * 2);             // transposed LN outputs (ping-pong with xT)
  float* P    = (float*)alloc((long)BATCH * CH * 196 * 4);
  bf16*  Uh   = (bf16*)alloc((long)BATCH * 784 * CH * 2);

  k_cvt<<<4096, 256, 0, stream>>>(x, xh, BNC);
  k_cvtXT<<<4096, 256, 0, stream>>>(x, xT);
  k_cvt<<<4096, 256, 0, stream>>>(w_fc1, wfcb, (long)CH * CH);
  k_cvt<<<4096, 256, 0, stream>>>(w1, w1b, (long)CH * CH);
  k_cvt<<<4096, 256, 0, stream>>>(w2, w2b, (long)CH * CH);
  k_cvt<<<4096, 256, 0, stream>>>(w3, w3b, (long)CH * CH);
  k_cvt_deconv<<<4096, 256, 0, stream>>>(dw, WhrT);

  const int MT = (NODES + TM - 1) / TM;  // 13
  // h = x @ w_fc1^T: B[k=c][n=d] = wfcb[d*CH+c]
  k_gemm<true><<<dim3(CH / TN, MT, BATCH), 256, 0, stream>>>(
      xh, BNC / BATCH, CH, wfcb, 0L, CH, hbuf, BNC / BATCH, CH, NODES, CH, CH);
  // s = h @ h^T: B[k=c][n=m] = h[m*CH+c]
  k_gemm<false><<<dim3(MT, MT, BATCH), 256, 0, stream>>>(
      hbuf, BNC / BATCH, CH, hbuf, BNC / BATCH, CH, sbuf, BNN / BATCH, NODES, NODES, NODES, CH);

  k_colsum<<<dim3((NODES + 255) / 256, BATCH), 256, 0, stream>>>(sbuf, cs);
  k_rowd<<<dim3(NODES, BATCH), 256, 0, stream>>>(sbuf, cs, dv);
  k_lap<<<8192, 256, 0, stream>>>(sbuf, cs, dv, laph);

  const float* gs[3] = {g1, g2, g3};
  const float* bs[3] = {b1, b2, b3};
  const bf16*  wbs[3] = {w1b, w2b, w3b};
  bf16* curT = xT;
  bf16* nxtT = cT1;
  for (int i = 0; i < 3; i++) {
    // g = lap @ cur: B[k=m][n=d] = curT[d*NODES+m]
    k_gemm<true><<<dim3(CH / TN, MT, BATCH), 256, 0, stream>>>(
        laph, BNN / BATCH, NODES, curT, (long)CH * NODES, NODES, gh, BNC / BATCH, CH, NODES, CH, NODES);
    // y = g @ wi^T: B[k=c][n=d] = wib[d*CH+c]
    k_gemm<false><<<dim3(CH / TN, MT, BATCH), 256, 0, stream>>>(
        gh, BNC / BATCH, CH, wbs[i], 0L, CH, yf, BNC / BATCH, CH, NODES, CH, CH);
    k_ln_leaky<<<BATCH * NODES, 256, 0, stream>>>(yf, gs[i], bs[i], nxtT);
    bf16* t = curT; curT = nxtT; nxtT = t;
  }

  k_conv1<<<(BATCH * 196 * CH) / 256, 256, 0, stream>>>(yf, cw, cb, P);
  k_upsample<<<(BATCH * 28 * 28 * CH) / 256, 256, 0, stream>>>(P, Uh);
  k_deconv<<<dim3(1024 / TN, 25, 9), 256, 0, stream>>>(Uh, WhrT, db, (float*)d_out);
}
